// Head_37263136260646
// MI455X (gfx1250) — matrix operand order repaired and checked by reading, not compiled, not run on hardware
//
#include <hip/hip_runtime.h>
#include <cstdint>
#include <cstddef>

// ---------------------------------------------------------------------------
// Types for CDNA5 WMMA (wave32)
// ---------------------------------------------------------------------------
typedef __bf16 bf16;
typedef __attribute__((ext_vector_type(16))) __bf16 v16bf;
typedef __attribute__((ext_vector_type(8)))  __bf16 v8bf;
typedef __attribute__((ext_vector_type(8)))  float  v8f;
typedef __attribute__((ext_vector_type(4)))  int    v4i;

typedef __attribute__((address_space(1))) v4i gas_v4i;  // global int4
typedef __attribute__((address_space(3))) v4i las_v4i;  // LDS int4

// Problem constants (match reference)
static constexpr int Bq = 4;
static constexpr int Sq = 2048;
static constexpr int Eq = 1024;
static constexpr int Hq = 1024;

// Staged LDS column stride (32 k-values + 8 pad) -> 80B: 16 lanes hit 16 banks
#define PADC 40

// --- gfx1250 async global->LDS path (guarded; sync fallback keeps it legal) --
#if defined(__has_builtin)
#if __has_builtin(__builtin_amdgcn_global_load_async_to_lds_b128)
#define HAVE_ASYNC_LDS 1
#endif
#endif

#if defined(HAVE_ASYNC_LDS)
#if __has_builtin(__builtin_amdgcn_s_wait_asynccnt)
#define WAIT_ASYNC() __builtin_amdgcn_s_wait_asynccnt(0)
#else
#define WAIT_ASYNC() asm volatile("s_wait_asynccnt 0" ::: "memory")
#endif
#else
#define WAIT_ASYNC() ((void)0)
#endif

// copy 32 contiguous bytes global -> LDS
__device__ __forceinline__ void stage_fill(const bf16* src, bf16* dst) {
#if defined(HAVE_ASYNC_LDS)
    __builtin_amdgcn_global_load_async_to_lds_b128(
        (gas_v4i*)src, (las_v4i*)dst, 0, 0);
    __builtin_amdgcn_global_load_async_to_lds_b128(
        (gas_v4i*)src, (las_v4i*)dst, 16, 0);
#else
    *reinterpret_cast<v8bf*>(dst)     = *reinterpret_cast<const v8bf*>(src);
    *reinterpret_cast<v8bf*>(dst + 8) = *reinterpret_cast<const v8bf*>(src + 8);
#endif
}

// ---------------------------------------------------------------------------
// fp32 -> bf16 conversion (elementwise)
// ---------------------------------------------------------------------------
__global__ void __launch_bounds__(256)
att_cvt_f32_bf16(const float* __restrict__ in, bf16* __restrict__ out, int n) {
    int i = blockIdx.x * 256 + threadIdx.x;
    if (i < n) out[i] = (bf16)in[i];
}

// ---------------------------------------------------------------------------
// WMMA fragment loads in this code's contiguous k order (k = 16*half + s; A and B agree)
//   A (16x32, MxK): lanes 0-15 row M=lane, elems K0..15 contiguous;
//                   lanes 16-31 row M=lane-16, elems K16..31
//   B (32x16, KxN): lanes 0-15 col N=lane, elems K0..15 contiguous;
//                   lanes 16-31 col N=lane-16, elems K16..31
//   C (16x16 f32): lane L col N=L%16; reg r -> row M = r + 8*(L/16)
// ---------------------------------------------------------------------------
__device__ __forceinline__ v16bf att_load_a(const bf16* A, int lda, int row,
                                            int k0, int half) {
    const bf16* p = A + (size_t)row * lda + k0 + half * 16;
    union { v16bf v; v8bf h[2]; } u;
    u.h[0] = *reinterpret_cast<const v8bf*>(p);
    u.h[1] = *reinterpret_cast<const v8bf*>(p + 8);
    return u.v;
}

// ---------------------------------------------------------------------------
// NT GEMM: C[m,n] = scale * sum_k A[m,k]*B[n,k] + bias[n]
//   - bf16 inputs, fp32 WMMA accumulate
//   - 256 threads = 8 waves: 4(M) x 2(N); wave tile 32x64; block tile 128x128
//   - B panel (128 cols x 32 k) staged in LDS via async global->LDS loads,
//     double buffered; fragments read back with ds_load_b128
//   - optional batched (z), causal tile skip, bf16/f32 store, transposed store
// ---------------------------------------------------------------------------
__global__ void __launch_bounds__(256)
att_gemm_nt_bf16(const bf16* __restrict__ A, const bf16* __restrict__ Bm,
                 const float* __restrict__ bias, void* __restrict__ C,
                 int N, int K, int lda, int ldb, int ldc,
                 long long sA, long long sB, long long sC,
                 float scale, int storeBf16, int causal, int transRows) {
    __shared__ bf16 smem[2][128 * PADC];

    const int z = blockIdx.z;
    const bf16* Ab = A + (size_t)z * sA;
    const bf16* Bb = Bm + (size_t)z * sB;

    const int rowBase = blockIdx.y * 128;
    const int colBase = blockIdx.x * 128;
    // causal: tile entirely above the diagonal -> never read by softmax
    if (causal && colBase > rowBase + 127) return;

    const int tid  = threadIdx.x;
    const int lane = tid & 31;
    const int wave = tid >> 5;
    const int wm   = wave & 3;   // 0..3 along M (32-row tiles)
    const int wn   = wave >> 2;  // 0..1 along N (64-col strips)
    const int lr   = lane & 15;
    const int half = lane >> 4;

    const int m0 = rowBase + wm * 32;
    const int n0 = wn * 64;                 // local col strip within block

    // cooperative fill mapping: thread -> (column, 16-element k-half)
    const int fcol  = tid >> 1;             // 0..127
    const int fhalf = tid & 1;              // 0..1
    const bf16* gfill = Bb + (size_t)(colBase + fcol) * ldb + fhalf * 16;
    bf16* lfill0 = &smem[0][fcol * PADC + fhalf * 16];
    bf16* lfill1 = &smem[1][fcol * PADC + fhalf * 16];

    v8f acc[2][4];
#pragma unroll
    for (int mt = 0; mt < 2; ++mt)
#pragma unroll
        for (int t = 0; t < 4; ++t)
#pragma unroll
            for (int i = 0; i < 8; ++i) acc[mt][t][i] = 0.0f;

    stage_fill(gfill, lfill0);              // prologue: stage k0 = 0

    int stage = 0;
    for (int k0 = 0; k0 < K; k0 += 32, stage ^= 1) {
        WAIT_ASYNC();
        __syncthreads();                    // stage `stage` is ready for all
        if (k0 + 32 < K)                    // prefetch next stage
            stage_fill(gfill + k0 + 32, stage ? lfill0 : lfill1);

        v16bf a0 = att_load_a(Ab, lda, m0 + lr,      k0, half);
        v16bf a1 = att_load_a(Ab, lda, m0 + 16 + lr, k0, half);
        const bf16* sbase = stage ? &smem[1][0] : &smem[0][0];
#pragma unroll
        for (int t = 0; t < 4; ++t) {
            const v16bf b = *reinterpret_cast<const v16bf*>(
                sbase + (n0 + t * 16 + lr) * PADC + half * 16);
            acc[0][t] = __builtin_amdgcn_wmma_f32_16x16x32_bf16(
                false, a0, false, b, (short)0, acc[0][t], false, false);
            acc[1][t] = __builtin_amdgcn_wmma_f32_16x16x32_bf16(
                false, a1, false, b, (short)0, acc[1][t], false, false);
        }
        __syncthreads();                    // all reads done before refill
    }

#pragma unroll
    for (int mt = 0; mt < 2; ++mt) {
#pragma unroll
        for (int t = 0; t < 4; ++t) {
            const int n = colBase + n0 + t * 16 + lr;
            const float bn = bias ? bias[n] : 0.0f;
#pragma unroll
            for (int r = 0; r < 8; ++r) {
                const int m = m0 + mt * 16 + half * 8 + r;
                const float v = acc[mt][t][r] * scale + bn;
                if (transRows > 0) {
                    // store C^T per batch-of-transRows rows: Ct[bb][n][sl]
                    const int bb = m / transRows;
                    const int sl = m - bb * transRows;
                    const size_t idx =
                        ((size_t)bb * N + n) * (size_t)transRows + sl;
                    ((bf16*)C)[idx] = (bf16)v;
                } else {
                    const size_t idx = (size_t)z * sC + (size_t)m * ldc + n;
                    if (storeBf16) ((bf16*)C)[idx] = (bf16)v;
                    else           ((float*)C)[idx] = v;
                }
            }
        }
    }
}

// ---------------------------------------------------------------------------
// RoPE in place on bf16 Q and K. One thread per (b,s) x (pair i < H/2).
// out[2i] = t[2i]*cos - t[2i+1]*sin ; out[2i+1] = t[2i+1]*cos + t[2i]*sin
// ---------------------------------------------------------------------------
__global__ void __launch_bounds__(256)
att_rope_inplace(bf16* __restrict__ Q, bf16* __restrict__ K,
                 const float* __restrict__ pos, int H) {
    const int halfH = H >> 1;
    const size_t idx = (size_t)blockIdx.x * 256 + threadIdx.x;
    const size_t bs = idx / halfH;
    const int i = (int)(idx - bs * halfH);

    const float p = pos[bs];
    const float omega = __powf(10000.0f, -(float)i / (float)halfH);
    float s, c;
    __sincosf(p * omega, &s, &c);

    const size_t base = bs * (size_t)H + 2 * (size_t)i;
    {
        const float q0 = (float)Q[base], q1 = (float)Q[base + 1];
        Q[base]     = (bf16)(q0 * c - q1 * s);
        Q[base + 1] = (bf16)(q1 * c + q0 * s);
    }
    {
        const float k0 = (float)K[base], k1 = (float)K[base + 1];
        K[base]     = (bf16)(k0 * c - k1 * s);
        K[base + 1] = (bf16)(k1 * c + k0 * s);
    }
}

// ---------------------------------------------------------------------------
// Causal softmax: one block per (q, b) row; reads fp32 scores (only k <= q),
// writes bf16 probabilities; masked columns get exact 0.
// ---------------------------------------------------------------------------
__global__ void __launch_bounds__(256)
att_softmax_causal(const float* __restrict__ scores, bf16* __restrict__ P,
                   int S) {
    const int q = blockIdx.x;
    const int b = blockIdx.y;
    const float* row = scores + ((size_t)b * S + q) * (size_t)S;
    bf16* prow = P + ((size_t)b * S + q) * (size_t)S;
    const int L = q + 1;

    __shared__ float red[256];
    const int tid = threadIdx.x;

    float mx = -3.4e38f;
    for (int c = tid; c < L; c += 256) mx = fmaxf(mx, row[c]);
    red[tid] = mx;
    __syncthreads();
    for (int off = 128; off > 0; off >>= 1) {
        if (tid < off) red[tid] = fmaxf(red[tid], red[tid + off]);
        __syncthreads();
    }
    mx = red[0];
    __syncthreads();

    float sum = 0.0f;
    for (int c = tid; c < L; c += 256) sum += __expf(row[c] - mx);
    red[tid] = sum;
    __syncthreads();
    for (int off = 128; off > 0; off >>= 1) {
        if (tid < off) red[tid] += red[tid + off];
        __syncthreads();
    }
    sum = red[0];
    __syncthreads();

    const float inv = 1.0f / sum;
    for (int c = tid; c < S; c += 256) {
        const float pv = (c < L) ? __expf(row[c] - mx) * inv : 0.0f;
        prow[c] = (bf16)pv;
    }
}

// ---------------------------------------------------------------------------
// Host-side launcher
// ---------------------------------------------------------------------------
extern "C" void kernel_launch(void* const* d_in, const int* in_sizes, int n_in,
                              void* d_out, int out_size, void* d_ws,
                              size_t ws_size, hipStream_t stream) {
    (void)in_sizes; (void)n_in; (void)out_size; (void)ws_size;

    const float* x   = (const float*)d_in[0];   // (B,S,E)
    const float* pos = (const float*)d_in[1];   // (B,S)
    // d_in[2] = mask (bool) — causality handled analytically
    const float* Wq = (const float*)d_in[3];
    const float* bq = (const float*)d_in[4];
    const float* Wk = (const float*)d_in[5];
    const float* bk = (const float*)d_in[6];
    const float* Wv = (const float*)d_in[7];
    const float* bv = (const float*)d_in[8];
    const float* Wo = (const float*)d_in[9];
    const float* bo = (const float*)d_in[10];
    float* out = (float*)d_out;                  // (B,S,H) fp32

    const int BS = Bq * Sq;                      // 8192
    char* ws = (char*)d_ws;
    size_t off = 0;
    auto alloc = [&](size_t bytes) -> char* {
        char* p = ws + off;
        off += (bytes + 255) & ~(size_t)255;
        return p;
    };

    bf16* xb  = (bf16*)alloc((size_t)BS * Eq * 2);       // x in bf16
    bf16* wqb = (bf16*)alloc((size_t)Hq * Eq * 2);
    bf16* wkb = (bf16*)alloc((size_t)Hq * Eq * 2);
    bf16* wvb = (bf16*)alloc((size_t)Hq * Eq * 2);
    bf16* wob = (bf16*)alloc((size_t)Hq * Hq * 2);
    bf16* qb  = (bf16*)alloc((size_t)BS * Hq * 2);       // roped Q
    bf16* kb  = (bf16*)alloc((size_t)BS * Hq * 2);       // roped K
    bf16* vt  = (bf16*)alloc((size_t)Bq * Hq * Sq * 2);  // V^T (B,H,S)
    float* sc = (float*)alloc((size_t)Bq * Sq * Sq * 4); // scores fp32
    bf16* pb  = (bf16*)alloc((size_t)Bq * Sq * Sq * 2);  // probs bf16
    bf16* attn = xb;  // alias: xb dead after the three projections

    // --- converts -----------------------------------------------------------
    {
        int n = BS * Eq;
        att_cvt_f32_bf16<<<(n + 255) / 256, 256, 0, stream>>>(x, xb, n);
        n = Hq * Eq;
        att_cvt_f32_bf16<<<(n + 255) / 256, 256, 0, stream>>>(Wq, wqb, n);
        att_cvt_f32_bf16<<<(n + 255) / 256, 256, 0, stream>>>(Wk, wkb, n);
        att_cvt_f32_bf16<<<(n + 255) / 256, 256, 0, stream>>>(Wv, wvb, n);
        att_cvt_f32_bf16<<<(n + 255) / 256, 256, 0, stream>>>(Wo, wob, n);
    }

    const dim3 blk(256);

    // --- Q/K/V projections (M=8192, N=1024, K=1024, NT) ---------------------
    {
        dim3 grid(Hq / 128, BS / 128, 1);
        att_gemm_nt_bf16<<<grid, blk, 0, stream>>>(
            xb, wqb, bq, qb, Hq, Eq, Eq, Eq, Hq, 0, 0, 0,
            1.0f, /*bf16*/1, /*causal*/0, /*trans*/0);
        att_gemm_nt_bf16<<<grid, blk, 0, stream>>>(
            xb, wkb, bk, kb, Hq, Eq, Eq, Eq, Hq, 0, 0, 0,
            1.0f, 1, 0, 0);
        // V projection stored transposed per batch: vt[b][h][s]
        att_gemm_nt_bf16<<<grid, blk, 0, stream>>>(
            xb, wvb, bv, vt, Hq, Eq, Eq, Eq, 0, 0, 0, 0,
            1.0f, 1, 0, /*transRows=*/Sq);
    }

    // --- RoPE on Q, K -------------------------------------------------------
    {
        size_t total = (size_t)BS * (Hq / 2);
        att_rope_inplace<<<(unsigned)(total / 256), 256, 0, stream>>>(
            qb, kb, pos, Hq);
    }

    // --- scores = Q K^T / 32 (batched, causal tile skip, fp32 out) ----------
    {
        dim3 grid(Sq / 128, Sq / 128, Bq);
        att_gemm_nt_bf16<<<grid, blk, 0, stream>>>(
            qb, kb, nullptr, sc, Sq, Hq, Hq, Hq, Sq,
            (long long)Sq * Hq, (long long)Sq * Hq, (long long)Sq * Sq,
            1.0f / 32.0f, /*bf16*/0, /*causal*/1, 0);
    }

    // --- masked softmax -> bf16 probs ---------------------------------------
    {
        dim3 grid(Sq, Bq);
        att_softmax_causal<<<grid, blk, 0, stream>>>(sc, pb, Sq);
    }

    // --- attn = P @ V  (A = P (S,S), B = V^T (H,S), NT) ---------------------
    {
        dim3 grid(Hq / 128, Sq / 128, Bq);
        att_gemm_nt_bf16<<<grid, blk, 0, stream>>>(
            pb, vt, nullptr, attn, Hq, Sq, Sq, Sq, Hq,
            (long long)Sq * Sq, (long long)Hq * Sq, (long long)Sq * Hq,
            1.0f, /*bf16*/1, 0, 0);
    }

    // --- output projection: out = attn @ Wo^T + bo (fp32 out) ---------------
    {
        dim3 grid(Hq / 128, BS / 128, 1);
        att_gemm_nt_bf16<<<grid, blk, 0, stream>>>(
            attn, wob, bo, out, Hq, Hq, Hq, Hq, Hq, 0, 0, 0,
            1.0f, /*bf16*/0, 0, 0);
    }
}
